// GDGNNModel_49881750175988
// MI455X (gfx1250) — matrix operand order repaired and checked by reading, not compiled, not run on hardware
//
#include <hip/hip_runtime.h>
#include <stdint.h>

// ================= CDNA5 (gfx1250) wave32 WMMA types =================
typedef __attribute__((ext_vector_type(16))) __bf16 v16bf;
typedef __attribute__((ext_vector_type(8)))  float  v8f;

union ABPack { v16bf v; uint32_t u[8]; };

__device__ __forceinline__ uint16_t f2bf(float f) {
  uint32_t u = __float_as_uint(f);
  u += 0x7FFFu + ((u >> 16) & 1u);   // round-to-nearest-even
  return (uint16_t)(u >> 16);
}

// low 32 bits of a generic pointer to __shared__ = LDS byte offset
__device__ __forceinline__ uint32_t lds_off32(const void* p) {
  return (uint32_t)(uintptr_t)p;
}

// ---------------- elementwise helpers ----------------
__global__ void k_zero(float* __restrict__ p, size_t n) {
  size_t i = (size_t)blockIdx.x * blockDim.x + threadIdx.x;
  size_t st = (size_t)gridDim.x * blockDim.x;
  for (; i < n; i += st) p[i] = 0.0f;
}

__global__ void k_f2bf(const float* __restrict__ s, uint16_t* __restrict__ d, size_t n) {
  size_t i = (size_t)blockIdx.x * blockDim.x + threadIdx.x;
  size_t st = (size_t)gridDim.x * blockDim.x;
  for (; i < n; i += st) d[i] = f2bf(s[i]);
}

// tew = concat(topic_vec[64x256], topic_edge_vec[64x512]) -> 64x768 bf16
__global__ void k_pack_tew(const float* __restrict__ tv, const float* __restrict__ tev,
                           uint16_t* __restrict__ tew) {
  int idx = blockIdx.x * blockDim.x + threadIdx.x;
  if (idx >= 64 * 768) return;
  int r = idx / 768, c = idx % 768;
  float v = (c < 256) ? tv[r * 256 + c] : tev[r * 512 + (c - 256)];
  tew[idx] = f2bf(v);
}

// ================= generic bf16 WMMA GEMM:  C[M,N] = A[M,K] @ B[N,K]^T =================
__global__ void k_gemm_bt(const uint16_t* __restrict__ A, int lda,
                          const uint16_t* __restrict__ Bm, int ldb,
                          float* __restrict__ C, int ldc,
                          int K, int ntM, int ntN) {
  int wid = blockIdx.x * (blockDim.x >> 5) + (threadIdx.x >> 5);
  if (wid >= ntM * ntN) return;                // wave-uniform exit (EXEC all-1 for WMMA)
  int lane = threadIdx.x & 31;
  int tM = wid / ntN, tN = wid % ntN;
  int lh = lane >> 4, l16 = lane & 15;
  const uint16_t* Ar = A + (size_t)(tM * 16 + l16) * lda;
  const uint16_t* Br = Bm + (size_t)(tN * 16 + l16) * ldb;
  v8f acc = {};
  for (int kb = 0; kb < K; kb += 32) {
    if (kb + 32 < K) __builtin_prefetch((const void*)(Br + kb + 32), 0, 3);
    ABPack a, b;
#pragma unroll
    for (int i = 0; i < 4; ++i) {              // A 16x32: K = 8*half+2i and 16+8*half+2i
      a.u[i]     = *(const uint32_t*)(Ar + kb + 8 * lh + 2 * i);
      a.u[4 + i] = *(const uint32_t*)(Ar + kb + 16 + 8 * lh + 2 * i);
    }
#pragma unroll
    for (int i = 0; i < 8; ++i)                // B 32x16: K = 8*half+2i (i<4) and 8+8*half+2i (i>=4)
      b.u[i] = *(const uint32_t*)(Br + kb + (i < 4 ? 8 * lh + 2 * i : 8 + 8 * lh + 2 * i));
    acc = __builtin_amdgcn_wmma_f32_16x16x32_bf16(false, a.v, false, b.v,
                                                  (short)0, acc, false, false);
  }
  float* Cp = C + (size_t)(tM * 16) * ldc + (size_t)tN * 16 + l16;
#pragma unroll
  for (int r = 0; r < 8; ++r)
    Cp[(size_t)(r + 8 * lh) * ldc] = acc[r];
}

// ========== fused gather-GEMM with async-LDS double buffering ==========
// be[64,Npad] = topic_edge_vec[64,512] @ ev^T ; ev rows gathered from wvb/noe.
// Per-wave pipeline: GLOBAL_LOAD_ASYNC_TO_LDS_B128 stages slab s+1 while WMMA
// consumes slab s; ASYNCcnt is per-wave and completes in order, so
// s_wait_asynccnt 4 / 0 is the only sync needed (no barriers).
__global__ void k_gemm_be(const uint16_t* __restrict__ Atev, // tew+256, lda=768
                          const uint16_t* __restrict__ wvb,  // V x 256 bf16
                          const uint16_t* __restrict__ noe,  // 512 bf16
                          const int* __restrict__ we0, const int* __restrict__ we1,
                          int EW, float* __restrict__ be, int ldc, int ntN) {
  // per wave: 2 bufs x (A 16*80B + B 16*80B) = 5120B; 4 waves/block
  __shared__ uint8_t lds_pipe[4 * 2 * 2560];
  int widb = threadIdx.x >> 5;
  int wid = blockIdx.x * (blockDim.x >> 5) + widb;
  if (wid >= 4 * ntN) return;
  int lane = threadIdx.x & 31;
  int tM = wid / ntN, tN = wid % ntN;
  int lh = lane >> 4, l16 = lane & 15;
  uint8_t* wbase = &lds_pipe[widb * 5120];

  // copy-role mapping: lane stages rows/cols {cr, cr+8}, 16B chunk c4
  int cr = lane >> 2, c4 = lane & 3;
  const uint8_t* aRow0 = (const uint8_t*)(Atev + (size_t)(tM * 16 + cr) * 768);
  const uint8_t* aRow1 = (const uint8_t*)(Atev + (size_t)(tM * 16 + cr + 8) * 768);
  const uint16_t* pl[2]; const uint16_t* ph[2];
#pragma unroll
  for (int j = 0; j < 2; ++j) {
    int col = tN * 16 + cr + 8 * j;            // ev row index for this staged column
    if (col == 0)       { pl[j] = noe;                           ph[j] = noe + 256; }
    else if (col <= EW) { pl[j] = wvb + (size_t)we0[col - 1] * 256;
                          ph[j] = wvb + (size_t)we1[col - 1] * 256; }
    else                { pl[j] = noe;                           ph[j] = noe + 256; }
  }

  auto stage = [&](int kb, int buf) {
    uint8_t* ab = wbase + buf * 2560;
    uint8_t* bb = ab + 1280;
    uint32_t lA0 = lds_off32(ab + cr * 80 + c4 * 16);
    uint32_t lA1 = lds_off32(ab + (cr + 8) * 80 + c4 * 16);
    uint32_t lB0 = lds_off32(bb + cr * 80 + c4 * 16);
    uint32_t lB1 = lds_off32(bb + (cr + 8) * 80 + c4 * 16);
    uint64_t gA0 = (uint64_t)(uintptr_t)(aRow0 + (size_t)kb * 2 + c4 * 16);
    uint64_t gA1 = (uint64_t)(uintptr_t)(aRow1 + (size_t)kb * 2 + c4 * 16);
    const uint16_t* b0 = (kb < 256) ? (pl[0] + kb) : (ph[0] + (kb - 256));
    const uint16_t* b1 = (kb < 256) ? (pl[1] + kb) : (ph[1] + (kb - 256));
    uint64_t gB0 = (uint64_t)(uintptr_t)((const uint8_t*)b0 + c4 * 16);
    uint64_t gB1 = (uint64_t)(uintptr_t)((const uint8_t*)b1 + c4 * 16);
    asm volatile("global_load_async_to_lds_b128 %0, %1, off" :: "v"(lA0), "v"(gA0) : "memory");
    asm volatile("global_load_async_to_lds_b128 %0, %1, off" :: "v"(lA1), "v"(gA1) : "memory");
    asm volatile("global_load_async_to_lds_b128 %0, %1, off" :: "v"(lB0), "v"(gB0) : "memory");
    asm volatile("global_load_async_to_lds_b128 %0, %1, off" :: "v"(lB1), "v"(gB1) : "memory");
  };

  v8f acc = {};
  stage(0, 0);
  const int S = 512 / 32;
  for (int s = 0; s < S; ++s) {
    int buf = s & 1;
    if (s + 1 < S) {
      stage((s + 1) * 32, buf ^ 1);
      asm volatile("s_wait_asynccnt 4" ::: "memory");  // slab s done; s+1 in flight
    } else {
      asm volatile("s_wait_asynccnt 0" ::: "memory");  // drain
    }
    const uint8_t* ab = wbase + buf * 2560;
    const uint8_t* bb = ab + 1280;
    ABPack a, b;
    const uint32_t* ap0 = (const uint32_t*)(ab + l16 * 80 + 16 * lh);
    const uint32_t* ap1 = (const uint32_t*)(ab + l16 * 80 + 32 + 16 * lh);
    const uint32_t* bp0 = (const uint32_t*)(bb + l16 * 80 + 16 * lh);
#pragma unroll
    for (int i = 0; i < 4; ++i) {
      a.u[i] = ap0[i]; a.u[4 + i] = ap1[i];
      b.u[i] = bp0[i]; b.u[4 + i] = bp0[8 + i];
    }
    acc = __builtin_amdgcn_wmma_f32_16x16x32_bf16(false, a.v, false, b.v,
                                                  (short)0, acc, false, false);
  }
  float* Cp = be + (size_t)(tM * 16) * ldc + (size_t)tN * 16 + l16;
#pragma unroll
  for (int r = 0; r < 8; ++r)
    Cp[(size_t)(r + 8 * lh) * ldc] = acc[r];
}

// ---------------- block reductions (blockDim == 256) ----------------
__device__ __forceinline__ float bred_sum(float v, float* sm) {
  int tid = threadIdx.x;
  __syncthreads(); sm[tid] = v; __syncthreads();
  for (int s = 128; s > 0; s >>= 1) { if (tid < s) sm[tid] += sm[tid + s]; __syncthreads(); }
  return sm[0];
}
__device__ __forceinline__ float bred_max(float v, float* sm) {
  int tid = threadIdx.x;
  __syncthreads(); sm[tid] = v; __syncthreads();
  for (int s = 128; s > 0; s >>= 1) { if (tid < s) sm[tid] = fmaxf(sm[tid], sm[tid + s]); __syncthreads(); }
  return sm[0];
}

// beta = softmax over rows (64 rows of length V)
__global__ void k_softmax_rows(float* __restrict__ X, int ncols) {
  __shared__ float sm[256];
  float* row = X + (size_t)blockIdx.x * ncols;
  int tid = threadIdx.x;
  float m = -3.4e38f;
  for (int j = tid; j < ncols; j += 256) m = fmaxf(m, row[j]);
  m = bred_max(m, sm);
  float s = 0.0f;
  for (int j = tid; j < ncols; j += 256) s += __expf(row[j] - m);
  s = bred_sum(s, sm);
  float inv = 1.0f / s;
  for (int j = tid; j < ncols; j += 256) row[j] = __expf(row[j] - m) * inv;
}

// l2-normalize head/tail halves of tv (64x64)
__global__ void k_tvnorm(const float* __restrict__ tv,
                         float* __restrict__ headn, float* __restrict__ tailn) {
  int i = threadIdx.x;
  if (i >= 64) return;
  const float* r = tv + i * 64;
  float h = 0.f, t = 0.f;
  for (int j = 0; j < 32; ++j) { h += r[j] * r[j]; t += r[32 + j] * r[32 + j]; }
  h = fmaxf(sqrtf(h), 1e-10f); t = fmaxf(sqrtf(t), 1e-10f);
  for (int j = 0; j < 32; ++j) {
    headn[i * 32 + j] = r[j] / h;
    tailn[i * 32 + j] = r[32 + j] / t;
  }
}

// W = clip(sigmoid(4*headn@tailn^T)); emit log(W)^T and log(1-W)^T in bf16 (GEMM B form)
__global__ void k_W(const float* __restrict__ headn, const float* __restrict__ tailn,
                    uint16_t* __restrict__ lpwT, uint16_t* __restrict__ lnwT) {
  int idx = blockIdx.x * blockDim.x + threadIdx.x;
  if (idx >= 64 * 64) return;
  int i = idx / 64, j = idx % 64;
  float s = 0.f;
  for (int c = 0; c < 32; ++c) s += headn[i * 32 + c] * tailn[j * 32 + c];
  float w = 1.0f / (1.0f + __expf(-4.0f * s));
  w = fminf(fmaxf(w, 1e-4f), 1.0f - 1e-4f);
  lpwT[j * 64 + i] = f2bf(logf(w));
  lnwT[j * 64 + i] = f2bf(logf(1.0f - w));
}

// node pass 1: batch counts, per-batch phi sums, KL2, gumbel argmax topic
__global__ void k_node1(const float* __restrict__ phi, const float* __restrict__ theta,
                        const float* __restrict__ idxw, const float* __restrict__ gum,
                        const int* __restrict__ xb, int N, int K,
                        float* __restrict__ cntB, float* __restrict__ Sb,
                        int* __restrict__ am, float* __restrict__ KL2a) {
  int i = blockIdx.x * blockDim.x + threadIdx.x;
  if (i >= N) return;
  int b = xb[i];
  atomicAdd(&cntB[b], 1.0f);
  const float* ph = phi + (size_t)i * K;
  const float* th = theta + (size_t)b * K;
  const float* gu = gum + (size_t)i * K;
  float kl = 0.f, best = -3.4e38f; int bi = 0;
  for (int k = 0; k < K; ++k) {
    float p = ph[k];
    kl += p * logf(p / (th[k] + 1e-10f) + 1e-10f);
    float t = logf(p) + gu[k];                 // argmax invariant to /TEMP
    if (t > best) { best = t; bi = k; }
    atomicAdd(&Sb[(size_t)b * K + k], p);
  }
  am[i] = bi;
  atomicAdd(&KL2a[b], idxw[i] * kl);
}

// node pass 2: n_edge closed form A[i].(S_b - phi[i]) and recon_word accumulation
__global__ void k_node2(const float* __restrict__ phi, const float* __restrict__ Amat,
                        const float* __restrict__ Sb, const float* __restrict__ beta, int V,
                        const float* __restrict__ idxw, const int* __restrict__ idx_x,
                        const int* __restrict__ xb, int N, int K,
                        float* __restrict__ neA, float* __restrict__ rwA) {
  int i = blockIdx.x * blockDim.x + threadIdx.x;
  if (i >= N) return;
  int b = xb[i], xi = idx_x[i];
  const float* Ai = Amat + (size_t)i * K;
  const float* ph = phi + (size_t)i * K;
  const float* sb = Sb + (size_t)b * K;
  float v = 0.f, rw = 0.f;
  for (int k = 0; k < K; ++k) {
    v += Ai[k] * (sb[k] - ph[k]);
    rw += ph[k] * logf(beta[(size_t)k * V + xi] + 1e-6f);
  }
  atomicAdd(&neA[b], v);
  atomicAdd(&rwA[b], idxw[i] * rw);
}

// edge pass 1: p_edge, NP histogram, same-batch edge subtraction of n_edge + NN count
__global__ void k_edge1(const float* __restrict__ phi, const float* __restrict__ Pm,
                        const float* __restrict__ Amat,
                        const int* __restrict__ e0, const int* __restrict__ e1,
                        const int* __restrict__ eib, const int* __restrict__ xb,
                        int E, int K,
                        float* __restrict__ peA, float* __restrict__ neA,
                        float* __restrict__ npA, float* __restrict__ nnE) {
  int e = blockIdx.x * blockDim.x + threadIdx.x;
  if (e >= E) return;
  int a = e0[e], c = e1[e], be_ = eib[e];
  const float* Pa = Pm + (size_t)a * K;
  const float* pc = phi + (size_t)c * K;
  float pe = 0.f;
  for (int k = 0; k < K; ++k) pe += Pa[k] * pc[k];
  atomicAdd(&peA[be_], pe);
  atomicAdd(&npA[be_], 1.0f);
  int ba = xb[a];
  if (ba == xb[c] && a != c) {
    const float* Aa = Amat + (size_t)a * K;
    float ne = 0.f;
    for (int k = 0; k < K; ++k) ne += Aa[k] * pc[k];
    atomicAdd(&neA[ba], -ne);
    atomicAdd(&nnE[ba], 1.0f);
  }
}

// logsumexp over columns of (be + log(bew+1e-20)); bew gathered from beta on the fly
__global__ void k_lse_be(const float* __restrict__ be, int ldbe,
                         const float* __restrict__ beta, int V,
                         const int* __restrict__ we0, const int* __restrict__ we1,
                         int EW, float* __restrict__ lse) {
  __shared__ float sm[256];
  int k = blockIdx.x, tid = threadIdx.x;
  const float* br = beta + (size_t)k * V;
  const float* ber = be + (size_t)k * ldbe;
  float s = 0.f;
  for (int j = tid; j < EW; j += 256) s += br[we0[j]] * br[we1[j]];
  s = bred_sum(s, sm);
  float bnw = fmaxf(1.0f - s + 1e-20f, 1e-20f);
  float m = (tid == 0) ? (ber[0] + logf(bnw)) : -3.4e38f;
  for (int j = tid; j < EW; j += 256) {
    float w = br[we0[j]] * br[we1[j]];
    m = fmaxf(m, ber[j + 1] + logf(w + 1e-20f));
  }
  m = bred_max(m, sm);
  float e = (tid == 0) ? __expf(ber[0] + logf(bnw) - m) : 0.f;
  for (int j = tid; j < EW; j += 256) {
    float w = br[we0[j]] * br[we1[j]];
    e += __expf(ber[j + 1] + logf(w + 1e-20f) - m);
  }
  e = bred_sum(e, sm);
  if (tid == 0) lse[k] = m + logf(e);
}

// edge pass 2: straight-through gumbel => one-hot argmax; scatter zg/besg
__global__ void k_edge2(const int* __restrict__ e0, const int* __restrict__ e1,
                        const int* __restrict__ am, const int* __restrict__ eid,
                        const int* __restrict__ eib, const float* __restrict__ edgew,
                        const float* __restrict__ be, int ldbe,
                        const float* __restrict__ lse, int E, int K,
                        float* __restrict__ zg, float* __restrict__ besg) {
  int e = blockIdx.x * blockDim.x + threadIdx.x;
  if (e >= E) return;
  int k0 = am[e0[e]];
  if (k0 != am[e1[e]]) return;                 // prod of one-hots is nonzero only if equal
  int b = eib[e];
  float w = edgew[e];
  float bs = __expf(be[(size_t)k0 * ldbe + eid[e]] - lse[k0]);
  atomicAdd(&zg[(size_t)b * K + k0], w);
  atomicAdd(&besg[(size_t)b * K + k0], bs * w);
}

__global__ void k_final(const float* __restrict__ KL1, const float* __restrict__ KL2a,
                        const float* __restrict__ peA, const float* __restrict__ neA,
                        const float* __restrict__ npA, const float* __restrict__ nnE,
                        const float* __restrict__ cntB, const float* __restrict__ rwA,
                        const float* __restrict__ zg, const float* __restrict__ besg,
                        int B, int K, float* __restrict__ out) {
  int b = blockIdx.x * blockDim.x + threadIdx.x;
  if (b >= B) return;
  float nb = cntB[b];
  float NN = nb * (nb - 1.0f) - nnE[b];
  float rs = -(peA[b] + neA[b] / (NN + 1e-6f) * npA[b]);
  float re = 0.f;
  for (int k = 0; k < K; ++k)
    re += logf(fmaxf(besg[(size_t)b * K + k], 1e-10f) / fmaxf(zg[(size_t)b * K + k], 1e-10f));
  out[b] = (-re) + (-rwA[b]) + KL1[b] + KL2a[b] + rs;
}

// ================================ launch ================================
extern "C" void kernel_launch(void* const* d_in, const int* in_sizes, int n_in,
                              void* d_out, int out_size, void* d_ws, size_t ws_size,
                              hipStream_t stream) {
  (void)n_in; (void)out_size;
  const float* phi   = (const float*)d_in[0];
  const float* theta = (const float*)d_in[1];
  const float* KL1   = (const float*)d_in[2];
  const float* idxw  = (const float*)d_in[3];
  const float* edgew = (const float*)d_in[4];
  const float* wvb   = (const float*)d_in[5];
  const float* tvec  = (const float*)d_in[6];
  const float* tevec = (const float*)d_in[7];
  const float* noe   = (const float*)d_in[8];
  const float* tlw   = (const float*)d_in[9];
  const float* gum   = (const float*)d_in[10];
  const int* idx_x   = (const int*)d_in[11];
  const int* xb      = (const int*)d_in[12];
  const int* ei      = (const int*)d_in[13];
  const int* eid     = (const int*)d_in[14];
  const int* eib     = (const int*)d_in[15];
  const int* we      = (const int*)d_in[16];

  const int K = 64, NI = 256;
  const int N  = in_sizes[0] / K;
  const int B  = in_sizes[2];
  const int E  = in_sizes[14];
  const int EW = in_sizes[16] / 2;
  const int V  = in_sizes[5] / NI;
  const int Npad = ((EW + 1 + 15) / 16) * 16;

  const int* e0 = ei;      const int* e1 = ei + E;
  const int* we0 = we;     const int* we1 = we + EW;

  char* p = (char*)d_ws;
  auto carve = [&](size_t bytes) -> void* {
    void* r = (void*)p; p += (bytes + 255) & ~(size_t)255; return r;
  };
  uint16_t* wvb_bf = (uint16_t*)carve((size_t)V * NI * 2);
  uint16_t* phi_bf = (uint16_t*)carve((size_t)N * K * 2);
  uint16_t* tew_bf = (uint16_t*)carve((size_t)64 * 768 * 2);
  uint16_t* tlw_bf = (uint16_t*)carve((size_t)64 * 768 * 2);
  uint16_t* noe_bf = (uint16_t*)carve((size_t)512 * 2);
  uint16_t* lpw_bf = (uint16_t*)carve((size_t)64 * 64 * 2);
  uint16_t* lnw_bf = (uint16_t*)carve((size_t)64 * 64 * 2);
  float* tv    = (float*)carve((size_t)64 * 64 * 4);
  float* headn = (float*)carve((size_t)64 * 32 * 4);
  float* tailn = (float*)carve((size_t)64 * 32 * 4);
  float* beta  = (float*)carve((size_t)K * V * 4);
  float* be    = (float*)carve((size_t)K * Npad * 4);
  float* lse   = (float*)carve((size_t)64 * 4);
  float* Pm    = (float*)carve((size_t)N * K * 4);
  float* Amat  = (float*)carve((size_t)N * K * 4);
  int*   am    = (int*)carve((size_t)N * 4);
  size_t naccf = (size_t)7 * B + (size_t)3 * B * K;
  float* acc   = (float*)carve(naccf * 4);
  if ((size_t)(p - (char*)d_ws) > ws_size) return;   // workspace too small: bail safely

  float* cntB = acc;
  float* KL2a = cntB + B;
  float* peA  = KL2a + B;
  float* neA  = peA + B;
  float* npA  = neA + B;
  float* nnE  = npA + B;
  float* rwA  = nnE + B;
  float* Sb   = rwA + B;
  float* zg   = Sb + (size_t)B * K;
  float* besg = zg + (size_t)B * K;

  auto gblk = [](size_t n) { size_t g = (n + 255) / 256; return (g > 16384) ? (size_t)16384 : g; };

  // 0) zero accumulators
  k_zero<<<dim3((unsigned)gblk(naccf)), dim3(256), 0, stream>>>(acc, naccf);
  // 1) bf16 conversions / packing
  k_f2bf<<<dim3((unsigned)gblk((size_t)V * NI)), dim3(256), 0, stream>>>(wvb, wvb_bf, (size_t)V * NI);
  k_f2bf<<<dim3((unsigned)gblk((size_t)N * K)), dim3(256), 0, stream>>>(phi, phi_bf, (size_t)N * K);
  k_f2bf<<<dim3((unsigned)gblk((size_t)64 * 768)), dim3(256), 0, stream>>>(tlw, tlw_bf, (size_t)64 * 768);
  k_f2bf<<<dim3(2), dim3(256), 0, stream>>>(noe, noe_bf, (size_t)512);
  k_pack_tew<<<dim3((64 * 768 + 255) / 256), dim3(256), 0, stream>>>(tvec, tevec, tew_bf);
  // 2) tv = tew @ topic_linear_w^T   (64x64, K=768)
  k_gemm_bt<<<dim3((4 * 4 + 3) / 4), dim3(128), 0, stream>>>(tew_bf, 768, tlw_bf, 768, tv, 64, 768, 4, 4);
  // 3) W chain -> log_PW^T / log_NW^T in bf16
  k_tvnorm<<<dim3(1), dim3(64), 0, stream>>>(tv, headn, tailn);
  k_W<<<dim3((64 * 64 + 255) / 256), dim3(256), 0, stream>>>(headn, tailn, lpw_bf, lnw_bf);
  // 4) beta logits = topic_vec @ wvb^T  (64 x V, K=256; A = tew cols [0,256) via lda=768)
  {
    int ntN = V / 16;
    k_gemm_bt<<<dim3((4 * ntN + 3) / 4), dim3(128), 0, stream>>>(tew_bf, 768, wvb_bf, 256, beta, V, 256, 4, ntN);
  }
  k_softmax_rows<<<dim3(64), dim3(256), 0, stream>>>(beta, V);
  // 5) P = Phi @ log_PW, A = Phi @ log_NW   (N x 64, K=64)
  {
    int ntM = N / 16;
    k_gemm_bt<<<dim3((ntM * 4 + 3) / 4), dim3(128), 0, stream>>>(phi_bf, 64, lpw_bf, 64, Pm, 64, 64, ntM, 4);
    k_gemm_bt<<<dim3((ntM * 4 + 3) / 4), dim3(128), 0, stream>>>(phi_bf, 64, lnw_bf, 64, Amat, 64, 64, ntM, 4);
  }
  // 6) be = topic_edge_vec @ ev^T, ev gathered on the fly, async double-buffered
  {
    int ntN = Npad / 16;
    k_gemm_be<<<dim3((4 * ntN + 3) / 4), dim3(128), 0, stream>>>(tew_bf + 256, wvb_bf, noe_bf, we0, we1, EW, be, Npad, ntN);
  }
  // 7) node / edge / row reductions
  k_node1<<<dim3((N + 255) / 256), dim3(256), 0, stream>>>(phi, theta, idxw, gum, xb, N, K, cntB, Sb, am, KL2a);
  k_lse_be<<<dim3(64), dim3(256), 0, stream>>>(be, Npad, beta, V, we0, we1, EW, lse);
  k_node2<<<dim3((N + 255) / 256), dim3(256), 0, stream>>>(phi, Amat, Sb, beta, V, idxw, idx_x, xb, N, K, neA, rwA);
  k_edge1<<<dim3((E + 255) / 256), dim3(256), 0, stream>>>(phi, Pm, Amat, e0, e1, eib, xb, E, K, peA, neA, npA, nnE);
  k_edge2<<<dim3((E + 255) / 256), dim3(256), 0, stream>>>(e0, e1, am, eid, eib, edgew, be, Npad, lse, E, K, zg, besg);
  // 8) combine per-graph loss
  k_final<<<dim3((B + 255) / 256), dim3(256), 0, stream>>>(KL1, KL2a, peA, neA, npA, nnE, cntB, rwA, zg, besg, B, K, (float*)d_out);
}
